// AttCo_85925115724075
// MI455X (gfx1250) — compile-verified, matrix operand order checked by reading, not run on hardware
//
#include <hip/hip_runtime.h>
#include <math.h>

// Problem constants
#define BB   4
#define CC   384
#define NN   1728      // 12^3
#define NH_  8
#define HD_  48
#define HDP  64        // padded head dim (zeros in 48..63 of q/k)

#define CSTAT_RB 8     // column-stats row chunks
#define CSTAT_TB 192   // column-stats threads (columns) per block; 1728 = 9*192

#define KSPLIT 27      // ctx GEMM split-K factor
#define KCHUNK (NN / KSPLIT)   // 64 = 2 WMMA K-steps

typedef unsigned short u16;
typedef unsigned int   u32;
typedef __attribute__((ext_vector_type(16))) __bf16 v16bf;
typedef __attribute__((ext_vector_type(8)))  float  v8f;

struct alignas(16) Frag { uint4 lo, hi; };

__device__ __forceinline__ u16 f2bf(float x) {
  u32 u = __builtin_bit_cast(u32, x);
  u += 0x7FFFu + ((u >> 16) & 1u);   // round-to-nearest-even
  return (u16)(u >> 16);
}

// Load a 16(row) x 32(K) bf16 fragment from a row-major buffer (ld in elements).
// ISA 7.12.2 layout: lanes 0-15 hold K=[0..7],[16..23]; lanes 16-31 hold K=[8..15],[24..31].
__device__ __forceinline__ v16bf load_frag(const u16* base, int row0, int k0, int ld) {
  const int lane = threadIdx.x & 31;
  const int r    = lane & 15;
  const int hsel = lane >> 4;
  const u16* p = base + (size_t)(row0 + r) * ld + (k0 + hsel * 8);
  Frag f;
  f.lo = *reinterpret_cast<const uint4*>(p);
  f.hi = *reinterpret_cast<const uint4*>(p + 16);
  return __builtin_bit_cast(v16bf, f);
}

__device__ __forceinline__ v8f wmma_bf16(v16bf a, v16bf b, v8f c) {
  return __builtin_amdgcn_wmma_f32_16x16x32_bf16(false, a, false, b, (short)0, c, false, false);
}

// f32 16x16 C/D layout: VGPR j -> row m0 + (lane>=16 ? 8 : 0) + j, col n0 + lane%16
__device__ __forceinline__ void store_f32_tile(float* out, int m0, int n0, int ld,
                                               v8f c, float scale) {
  const int lane = threadIdx.x & 31;
  const int col  = lane & 15;
  const int rb   = (lane >> 4) * 8;
  float* p = out + (size_t)(m0 + rb) * ld + n0 + col;
#pragma unroll
  for (int j = 0; j < 8; ++j) p[(size_t)j * ld] = c[j] * scale;
}

// ---------------- utility kernels ----------------

__global__ void zero_kernel(u32* p, size_t n) {
  size_t i = (size_t)blockIdx.x * blockDim.x + threadIdx.x;
  if (i < n) p[i] = 0u;
}

// x [B,C,N] f32  ->  h [B,N,C] bf16
__global__ void transpose_convert(const float* __restrict__ x, u16* __restrict__ h) {
  size_t i = (size_t)blockIdx.x * blockDim.x + threadIdx.x;
  const size_t total = (size_t)BB * NN * CC;
  if (i >= total) return;
  int c = (int)(i % CC);
  size_t t = i / CC;
  int n = (int)(t % NN);
  int b = (int)(t / NN);
  h[i] = f2bf(x[((size_t)b * CC + c) * NN + n]);
}

__global__ void convert_kernel(const float* __restrict__ src, u16* __restrict__ dst, int n) {
  int i = blockIdx.x * blockDim.x + threadIdx.x;
  if (i < n) dst[i] = f2bf(src[i]);
}

// ---------------- projection GEMM (32x32 per wave) ----------------
// y[b,n,co] = sum_ci h[b,n,ci] * W[co,ci] + bias[co]
// mode 0: store bf16 to out[((b*NH+head)*N + n)*HDP + d]   (q,k; padded layout)
// mode 1: store bf16 to out[((b*NH+head)*HDP + d)*N + n]   (v transposed)
__global__ void proj_kernel(const u16* __restrict__ h, const u16* __restrict__ w,
                            const float* __restrict__ bias, u16* __restrict__ out,
                            int mode) {
  const int b  = blockIdx.z;
  const int nt = blockIdx.x * 32;
  const int ct = blockIdx.y * 32;
  const u16* A = h + (size_t)b * NN * CC;
  v8f acc[2][2] = {};
  for (int kk = 0; kk < CC; kk += 32) {
    v16bf a0 = load_frag(A, nt,      kk, CC);
    v16bf a1 = load_frag(A, nt + 16, kk, CC);
    v16bf b0 = load_frag(w, ct,      kk, CC);
    v16bf b1 = load_frag(w, ct + 16, kk, CC);
    acc[0][0] = wmma_bf16(a0, b0, acc[0][0]);
    acc[0][1] = wmma_bf16(a0, b1, acc[0][1]);
    acc[1][0] = wmma_bf16(a1, b0, acc[1][0]);
    acc[1][1] = wmma_bf16(a1, b1, acc[1][1]);
  }
  const int lane = threadIdx.x & 31;
  const int col  = lane & 15;
  const int rb   = (lane >> 4) * 8;
#pragma unroll
  for (int i = 0; i < 2; ++i) {
#pragma unroll
    for (int jt = 0; jt < 2; ++jt) {
      const int co   = ct + jt * 16 + col;
      const int head = co / HD_;
      const int d    = co - head * HD_;
      const float bv = bias[co];
#pragma unroll
      for (int j = 0; j < 8; ++j) {
        const int n = nt + i * 16 + rb + j;
        const u16 v = f2bf(acc[i][jt][j] + bv);
        if (mode == 0) out[(((size_t)b * NH_ + head) * NN + n) * HDP + d] = v;
        else           out[(((size_t)b * NH_ + head) * HDP + d) * NN + n] = v;
      }
    }
  }
}

// ---------------- scores: S[q,k] = scale * q . k  (32x64 per wave) ----------------
// q,k pre-offset to (b,h): [N, HDP] bf16.
__global__ void scores_kernel(const u16* __restrict__ q, const u16* __restrict__ k,
                              float* __restrict__ S, float scale) {
  const int qt = blockIdx.x * 32;
  const int kt = blockIdx.y * 64;
  v8f acc[2][4] = {};
  for (int kk = 0; kk < HDP; kk += 32) {
    v16bf a0 = load_frag(q, qt,      kk, HDP);
    v16bf a1 = load_frag(q, qt + 16, kk, HDP);
    v16bf bf[4];
#pragma unroll
    for (int j = 0; j < 4; ++j) bf[j] = load_frag(k, kt + j * 16, kk, HDP);
#pragma unroll
    for (int j = 0; j < 4; ++j) {
      acc[0][j] = wmma_bf16(a0, bf[j], acc[0][j]);
      acc[1][j] = wmma_bf16(a1, bf[j], acc[1][j]);
    }
  }
#pragma unroll
  for (int i = 0; i < 2; ++i)
#pragma unroll
    for (int j = 0; j < 4; ++j)
      store_f32_tile(S, qt + i * 16, kt + j * 16, NN, acc[i][j], scale);
}

// ---------------- softmax stats ----------------
// Row stats: one 256-thread block per row, LDS tree reduction.
__global__ void row_stats(const float* __restrict__ S, float* __restrict__ rmax,
                          float* __restrict__ rsum) {
  __shared__ float red[256];
  const int q = blockIdx.x;
  const int t = threadIdx.x;
  const float* row = S + (size_t)q * NN;
  float m = -3.4e38f;
  for (int j = t; j < NN; j += 256) m = fmaxf(m, row[j]);
  red[t] = m; __syncthreads();
  for (int s2 = 128; s2 > 0; s2 >>= 1) {
    if (t < s2) red[t] = fmaxf(red[t], red[t + s2]);
    __syncthreads();
  }
  m = red[0]; __syncthreads();
  float s = 0.f;
  for (int j = t; j < NN; j += 256) s += __expf(row[j] - m);
  red[t] = s; __syncthreads();
  for (int s2 = 128; s2 > 0; s2 >>= 1) {
    if (t < s2) red[t] += red[t + s2];
    __syncthreads();
  }
  if (t == 0) { rmax[q] = m; rsum[q] = red[0]; }
}

// Column stats pass 1: partial (max, expsum) per (row-chunk, column). Coalesced column reads.
__global__ void col_stats_partial(const float* __restrict__ S, float* __restrict__ pm,
                                  float* __restrict__ ps) {
  const int ki = blockIdx.x * CSTAT_TB + threadIdx.x;
  const int rc = blockIdx.y;
  const int rows = NN / CSTAT_RB;   // 216
  const int r0 = rc * rows;
  float m = -3.4e38f;
  for (int j = 0; j < rows; ++j) m = fmaxf(m, S[(size_t)(r0 + j) * NN + ki]);
  float s = 0.f;
  for (int j = 0; j < rows; ++j) s += __expf(S[(size_t)(r0 + j) * NN + ki] - m);
  pm[(size_t)rc * NN + ki] = m;
  ps[(size_t)rc * NN + ki] = s;
}

// Column stats pass 2: streaming-softmax combine across row chunks.
__global__ void col_stats_combine(const float* __restrict__ pm, const float* __restrict__ ps,
                                  float* __restrict__ cmax, float* __restrict__ csum) {
  int ki = blockIdx.x * blockDim.x + threadIdx.x;
  if (ki >= NN) return;
  float m = -3.4e38f;
#pragma unroll
  for (int r = 0; r < CSTAT_RB; ++r) m = fmaxf(m, pm[(size_t)r * NN + ki]);
  float s = 0.f;
#pragma unroll
  for (int r = 0; r < CSTAT_RB; ++r)
    s += ps[(size_t)r * NN + ki] * __expf(pm[(size_t)r * NN + ki] - m);
  cmax[ki] = m; csum[ki] = s;
}

// P1[q,k] = softmax over q (column), row-major.  P2T[k,q] = softmax over k (row), transposed.
__global__ void softmax_kernel(const float* __restrict__ S,
                               const float* __restrict__ rmax, const float* __restrict__ rsum,
                               const float* __restrict__ cmax, const float* __restrict__ csum,
                               u16* __restrict__ P1, u16* __restrict__ P2T) {
  size_t i = (size_t)blockIdx.x * blockDim.x + threadIdx.x;
  if (i >= (size_t)NN * NN) return;
  int ki = (int)(i % NN);
  int qi = (int)(i / NN);
  float s = S[i];
  P1[i] = f2bf(__expf(s - cmax[ki]) / csum[ki]);
  P2T[(size_t)ki * NN + qi] = f2bf(__expf(s - rmax[qi]) / rsum[qi]);
}

// ---------------- context GEMM, split-K (32x48 per wave, K chunk = 64) ----------------
// pc[ks][n][d] = sum_{k in chunk ks} P[n,k] * vT[d,k]
__global__ void ctx_partial_kernel(const u16* __restrict__ P, const u16* __restrict__ vT,
                                   float* __restrict__ pc) {
  const int nt = blockIdx.x * 32;
  const int ks = blockIdx.y;
  const int k0 = ks * KCHUNK;
  v8f acc[2][3] = {};
  for (int kk = k0; kk < k0 + KCHUNK; kk += 32) {
    v16bf a0 = load_frag(P, nt,      kk, NN);
    v16bf a1 = load_frag(P, nt + 16, kk, NN);
    v16bf bf[3];
#pragma unroll
    for (int j = 0; j < 3; ++j) bf[j] = load_frag(vT, j * 16, kk, NN);
#pragma unroll
    for (int j = 0; j < 3; ++j) {
      acc[0][j] = wmma_bf16(a0, bf[j], acc[0][j]);
      acc[1][j] = wmma_bf16(a1, bf[j], acc[1][j]);
    }
  }
  float* slab = pc + (size_t)ks * NN * HD_;
  const int lane = threadIdx.x & 31;
  const int col  = lane & 15;
  const int rb   = (lane >> 4) * 8;
#pragma unroll
  for (int i = 0; i < 2; ++i)
#pragma unroll
    for (int jt = 0; jt < 3; ++jt) {
      const int d = jt * 16 + col;
#pragma unroll
      for (int j = 0; j < 8; ++j) {
        const int n = nt + i * 16 + rb + j;
        slab[(size_t)n * HD_ + d] = acc[i][jt][j];
      }
    }
}

// Reduce split-K partials -> ctx[b,n,head*48+d] (bf16)
__global__ void ctx_reduce_kernel(const float* __restrict__ pc, u16* __restrict__ ctx,
                                  int b, int head) {
  int i = blockIdx.x * blockDim.x + threadIdx.x;   // over NN*HD_
  if (i >= NN * HD_) return;
  const int d = i % HD_;
  const int n = i / HD_;
  float s = 0.f;
  for (int ks = 0; ks < KSPLIT; ++ks) s += pc[(size_t)ks * NN * HD_ + i];
  ctx[((size_t)b * NN + n) * CC + head * HD_ + d] = f2bf(s);
}

// ---------------- output projection + bias + residual (32x32 per wave) ----------------
// out[b,co,n] = sum_ci ctx[b,n,ci]*W[co,ci] + bias[co] + x[b,co,n]
__global__ void oproj_kernel(const u16* __restrict__ ctx, const u16* __restrict__ w,
                             const float* __restrict__ bias, const float* __restrict__ x,
                             float* __restrict__ out) {
  const int b  = blockIdx.z;
  const int nt = blockIdx.x * 32;
  const int ct = blockIdx.y * 32;
  const u16* A = ctx + (size_t)b * NN * CC;
  v8f acc[2][2] = {};
  for (int kk = 0; kk < CC; kk += 32) {
    v16bf a0 = load_frag(A, nt,      kk, CC);
    v16bf a1 = load_frag(A, nt + 16, kk, CC);
    v16bf b0 = load_frag(w, ct,      kk, CC);
    v16bf b1 = load_frag(w, ct + 16, kk, CC);
    acc[0][0] = wmma_bf16(a0, b0, acc[0][0]);
    acc[0][1] = wmma_bf16(a0, b1, acc[0][1]);
    acc[1][0] = wmma_bf16(a1, b0, acc[1][0]);
    acc[1][1] = wmma_bf16(a1, b1, acc[1][1]);
  }
  const int lane = threadIdx.x & 31;
  const int col  = lane & 15;
  const int rb   = (lane >> 4) * 8;
#pragma unroll
  for (int i = 0; i < 2; ++i) {
#pragma unroll
    for (int jt = 0; jt < 2; ++jt) {
      const int co   = ct + jt * 16 + col;
      const float bv = bias[co];
#pragma unroll
      for (int j = 0; j < 8; ++j) {
        const int n = nt + i * 16 + rb + j;
        const size_t idx = ((size_t)b * CC + co) * NN + n;
        out[idx] = acc[i][jt][j] + bv + x[idx];
      }
    }
  }
}

// ---------------- host ----------------

extern "C" void kernel_launch(void* const* d_in, const int* in_sizes, int n_in,
                              void* d_out, int out_size, void* d_ws, size_t ws_size,
                              hipStream_t stream) {
  (void)in_sizes; (void)n_in; (void)out_size; (void)ws_size;
  const float* x1 = (const float*)d_in[0];
  const float* x2 = (const float*)d_in[1];
  const float* Wsrc[6] = {(const float*)d_in[2], (const float*)d_in[3], (const float*)d_in[4],
                          (const float*)d_in[5], (const float*)d_in[6], (const float*)d_in[7]};
  const float* bq  = (const float*)d_in[8];
  const float* bk  = (const float*)d_in[9];
  const float* bv1 = (const float*)d_in[10];
  const float* bv2 = (const float*)d_in[11];
  const float* bo1 = (const float*)d_in[12];
  const float* bo2 = (const float*)d_in[13];
  float* out = (float*)d_out;

  // workspace layout (bytes, all 256-aligned)
  const size_t SZ_H   = (size_t)BB * NN * CC * 2;        // 5,308,416
  const size_t SZ_W   = (size_t)6 * CC * CC * 2;         // 1,769,472
  const size_t SZ_QK  = (size_t)BB * NH_ * NN * HDP * 2; // 7,077,888
  const size_t SZ_S   = (size_t)NN * NN * 4;             // 11,943,936
  const size_t SZ_P   = (size_t)NN * NN * 2;             // 5,971,968
  const size_t SZ_PC  = (size_t)KSPLIT * NN * HD_ * 4;   // 8,957,952
  char* ws = (char*)d_ws;
  size_t off = 0;
  u16*   h1   = (u16*)(ws + off); off += SZ_H;
  u16*   h2   = (u16*)(ws + off); off += SZ_H;
  u16*   wbf  = (u16*)(ws + off); off += SZ_W;
  u16*   qb   = (u16*)(ws + off); off += SZ_QK;
  u16*   kb   = (u16*)(ws + off); off += SZ_QK;
  u16*   v1T  = (u16*)(ws + off); off += SZ_QK;
  u16*   v2T  = (u16*)(ws + off); off += SZ_QK;
  u16*   ctx1 = (u16*)(ws + off); off += SZ_H;
  u16*   ctx2 = (u16*)(ws + off); off += SZ_H;
  float* Sbuf = (float*)(ws + off); off += SZ_S;
  u16*   P1   = (u16*)(ws + off); off += SZ_P;
  u16*   P2T  = (u16*)(ws + off); off += SZ_P;
  float* pcbuf= (float*)(ws + off); off += SZ_PC;
  float* rmax = (float*)(ws + off); off += NN * 4;
  float* rsum = (float*)(ws + off); off += NN * 4;
  float* cmax = (float*)(ws + off); off += NN * 4;
  float* csum = (float*)(ws + off); off += NN * 4;
  float* pm   = (float*)(ws + off); off += (size_t)CSTAT_RB * NN * 4;
  float* ps   = (float*)(ws + off); off += (size_t)CSTAT_RB * NN * 4;

  const u16* wq  = wbf + (size_t)0 * CC * CC;
  const u16* wk  = wbf + (size_t)1 * CC * CC;
  const u16* wv1 = wbf + (size_t)2 * CC * CC;
  const u16* wv2 = wbf + (size_t)3 * CC * CC;
  const u16* wo1 = wbf + (size_t)4 * CC * CC;
  const u16* wo2 = wbf + (size_t)5 * CC * CC;

  // zero padded q/k buffers (d = 48..63 must be zero for the K=64 scores GEMM)
  {
    size_t nzero = (2 * SZ_QK) / 4;
    zero_kernel<<<dim3((unsigned)((nzero + 255) / 256)), 256, 0, stream>>>((u32*)qb, nzero);
  }

  // convert inputs and weights to bf16
  {
    size_t tot = (size_t)BB * NN * CC;
    transpose_convert<<<dim3((unsigned)((tot + 255) / 256)), 256, 0, stream>>>(x1, h1);
    transpose_convert<<<dim3((unsigned)((tot + 255) / 256)), 256, 0, stream>>>(x2, h2);
    for (int i = 0; i < 6; ++i)
      convert_kernel<<<dim3((CC * CC + 255) / 256), 256, 0, stream>>>(
          Wsrc[i], wbf + (size_t)i * CC * CC, CC * CC);
  }

  // projections (32x32 tiles)
  dim3 pgrid(NN / 32, CC / 32, BB);
  proj_kernel<<<pgrid, 32, 0, stream>>>(h1, wq,  bq,  qb,  0);
  proj_kernel<<<pgrid, 32, 0, stream>>>(h2, wk,  bk,  kb,  0);
  proj_kernel<<<pgrid, 32, 0, stream>>>(h1, wv1, bv1, v1T, 1);
  proj_kernel<<<pgrid, 32, 0, stream>>>(h2, wv2, bv2, v2T, 1);

  const float scale = 1.0f / sqrtf((float)HD_);
  const dim3 cgrid(NN / 32, KSPLIT);
  const dim3 rgrid((NN * HD_ + 255) / 256);
  for (int b = 0; b < BB; ++b) {
    for (int hh = 0; hh < NH_; ++hh) {
      const size_t bh = (size_t)b * NH_ + hh;
      const u16* qbh  = qb  + bh * NN * HDP;
      const u16* kbh  = kb  + bh * NN * HDP;
      const u16* v1bh = v1T + bh * HDP * NN;
      const u16* v2bh = v2T + bh * HDP * NN;

      scores_kernel<<<dim3(NN / 32, NN / 64), 32, 0, stream>>>(qbh, kbh, Sbuf, scale);
      row_stats<<<dim3(NN), 256, 0, stream>>>(Sbuf, rmax, rsum);
      col_stats_partial<<<dim3(NN / CSTAT_TB, CSTAT_RB), CSTAT_TB, 0, stream>>>(Sbuf, pm, ps);
      col_stats_combine<<<dim3((NN + 255) / 256), 256, 0, stream>>>(pm, ps, cmax, csum);
      softmax_kernel<<<dim3((unsigned)(((size_t)NN * NN + 255) / 256)), 256, 0, stream>>>(
          Sbuf, rmax, rsum, cmax, csum, P1, P2T);

      ctx_partial_kernel<<<cgrid, 32, 0, stream>>>(P1, v1bh, pcbuf);
      ctx_reduce_kernel<<<rgrid, 256, 0, stream>>>(pcbuf, ctx1, b, hh);
      ctx_partial_kernel<<<cgrid, 32, 0, stream>>>(P2T, v2bh, pcbuf);
      ctx_reduce_kernel<<<rgrid, 256, 0, stream>>>(pcbuf, ctx2, b, hh);
    }
  }

  dim3 ogrid(NN / 32, CC / 32, BB);
  oproj_kernel<<<ogrid, 32, 0, stream>>>(ctx1, wo1, bo1, x1, out);
  oproj_kernel<<<ogrid, 32, 0, stream>>>(ctx2, wo2, bo2, x2, out + (size_t)BB * CC * NN);
}
